// GPT_91147795955961
// MI455X (gfx1250) — compile-verified, matrix operand order checked by reading, not run on hardware
//
#include <hip/hip_runtime.h>
#include <hip/hip_bf16.h>
#include <cmath>

typedef __bf16 bf16_t;
typedef __attribute__((ext_vector_type(8)))  bf16_t v8bf;
typedef __attribute__((ext_vector_type(16))) bf16_t v16bf;
typedef __attribute__((ext_vector_type(8)))  float  v8f;

static constexpr int Bk = 4, Tk = 1024, Ck = 768, Hk = 12, Lk = 6, Vk = 50257, Dk = 64;

// ---- async global->LDS (CDNA5) feature detection, with safe fallback ------
#if defined(__has_builtin)
#if __has_builtin(__builtin_amdgcn_global_load_async_to_lds_b128)
#define USE_ASYNC_LDS 1
#endif
#endif
#ifndef USE_ASYNC_LDS
#define USE_ASYNC_LDS 0
#endif

#if USE_ASYNC_LDS
typedef int v4i __attribute__((vector_size(16)));
typedef __attribute__((address_space(1))) v4i* g4p;   // global int4*
typedef __attribute__((address_space(3))) v4i* l4p;   // LDS int4*
#define ASYNC_CP_B128(g, l, off) \
  __builtin_amdgcn_global_load_async_to_lds_b128((g4p)(g), (l4p)(l), (off), 0)
#if __has_builtin(__builtin_amdgcn_s_wait_asynccnt)
#define WAIT_ASYNC() __builtin_amdgcn_s_wait_asynccnt(0)
#else
#define WAIT_ASYNC() asm volatile("s_wait_asynccnt 0" ::: "memory")
#endif
#endif

__device__ __forceinline__ bf16_t f2bf(float f) {
  unsigned u = __builtin_bit_cast(unsigned, f);
  unsigned r = u + 0x7FFFu + ((u >> 16) & 1u);           // round-to-nearest-even
  return __builtin_bit_cast(bf16_t, (unsigned short)(r >> 16));
}
__device__ __forceinline__ unsigned pack2f(float lo, float hi) {
  return (unsigned)__builtin_bit_cast(unsigned short, f2bf(lo)) |
         ((unsigned)__builtin_bit_cast(unsigned short, f2bf(hi)) << 16);
}
__device__ __forceinline__ unsigned packbf(bf16_t lo, bf16_t hi) {
  return (unsigned)__builtin_bit_cast(unsigned short, lo) |
         ((unsigned)__builtin_bit_cast(unsigned short, hi) << 16);
}

// Load one 16-bf16 A/B fragment for v_wmma_f32_16x16x32_bf16.
// p = &tile[row*stride + kbase + (lane>>4)*8]; two contiguous 16-byte runs
// (K = kbase+hi*8+{0..7} and K = kbase+16+hi*8+{0..7}) per CDNA5 layout.
__device__ __forceinline__ v16bf load_frag(const bf16_t* p) {
  v8bf lo = *(const v8bf*)p;
  v8bf hi = *(const v8bf*)(p + 16);
  return __builtin_shufflevector(lo, hi, 0,1,2,3,4,5,6,7,8,9,10,11,12,13,14,15);
}

// ---------------------------------------------------------------- embedding
__global__ __launch_bounds__(256) void embed_kernel(
    const int* __restrict__ tok, const float* __restrict__ wte,
    const float* __restrict__ wpe, float* __restrict__ h) {
  const int row = blockIdx.x;                 // b*T + t
  const int t   = row & (Tk - 1);
  const int tkn = tok[row];
  for (int c = threadIdx.x; c < Ck; c += 256)
    h[(size_t)row * Ck + c] = wte[(size_t)tkn * Ck + c] + wpe[(size_t)t * Ck + c];
}

// ---------------------------------------------------------------- layernorm
__global__ __launch_bounds__(256) void layernorm_kernel(
    const float* __restrict__ x, const float* __restrict__ g,
    const float* __restrict__ bta, bf16_t* __restrict__ out) {
  __shared__ float red[8];
  const int row = blockIdx.x, tid = threadIdx.x;
  const float* xr = x + (size_t)row * Ck;
  float v0 = xr[tid], v1 = xr[tid + 256], v2 = xr[tid + 512];
  float s = v0 + v1 + v2;
  #pragma unroll
  for (int off = 16; off; off >>= 1) s += __shfl_xor(s, off);
  if ((tid & 31) == 0) red[tid >> 5] = s;
  __syncthreads();
  float mean = 0.f;
  #pragma unroll
  for (int i = 0; i < 8; ++i) mean += red[i];
  mean *= (1.0f / Ck);
  __syncthreads();
  const float d0 = v0 - mean, d1 = v1 - mean, d2 = v2 - mean;
  s = d0 * d0 + d1 * d1 + d2 * d2;
  #pragma unroll
  for (int off = 16; off; off >>= 1) s += __shfl_xor(s, off);
  if ((tid & 31) == 0) red[tid >> 5] = s;
  __syncthreads();
  float var = 0.f;
  #pragma unroll
  for (int i = 0; i < 8; ++i) var += red[i];
  var *= (1.0f / Ck);
  const float rstd = rsqrtf(var + 1e-5f);
  bf16_t* o = out + (size_t)row * Ck;
  o[tid]       = f2bf(d0 * rstd * g[tid]       + bta[tid]);
  o[tid + 256] = f2bf(d1 * rstd * g[tid + 256] + bta[tid + 256]);
  o[tid + 512] = f2bf(d2 * rstd * g[tid + 512] + bta[tid + 512]);
}

// ---------------------------------------------------------------- WMMA GEMM
// out = act( A[M,K](bf16) @ W[K,N](f32) + bias ) (+ resid), A row-major.
// BM=128 BN=128 BK=64, 256 threads = 8 waves, wave -> 32x64 (2x4 WMMA tiles).
// A tile: async global->LDS byte copy; W tile: fp32->bf16 packed u32 stores.
template<bool BIAS, bool GELU, bool RES, bool OUTBF>
__global__ __launch_bounds__(256) void gemm_bf16_kernel(
    const bf16_t* __restrict__ A, const float* __restrict__ W,
    const float* __restrict__ bias, const float* resid,
    float* outF, bf16_t* outB, int M, int N, int K) {
  constexpr int BKt = 64;
  __shared__ bf16_t sA[128 * BKt];           // [m][k]
  __shared__ bf16_t sB[128 * BKt];           // transposed: [n][k]

  const int tid  = threadIdx.x;
  const int lane = tid & 31, wave = tid >> 5;
  const int lm = lane & 15, hb = lane >> 4;
  const int wr = wave >> 1, wc = wave & 1;
  const int m0 = blockIdx.y * 128, n0 = blockIdx.x * 128;

  v8f acc[2][4];
  const v8f zv = {};
  #pragma unroll
  for (int i = 0; i < 2; ++i)
    #pragma unroll
    for (int j = 0; j < 4; ++j) acc[i][j] = zv;

  const int arow = tid >> 1, aks = (tid & 1) * 32;   // A: 32 bf16 per thread
  const int wkp  = tid >> 3, wns = (tid & 7) * 16;   // W: k-pair x 16 n per thread
  // clamp OOB A rows to a valid row; their outputs are discarded in epilogue
  const int arc = (m0 + arow < M) ? (m0 + arow) : (M - 1);

  for (int kt = 0; kt < K; kt += BKt) {
    {
      const bf16_t* src = A + (size_t)arc * K + kt + aks;
      bf16_t* dst = &sA[arow * BKt + aks];
#if USE_ASYNC_LDS
      ASYNC_CP_B128(src, dst, 0);
      ASYNC_CP_B128(src, dst, 16);
      ASYNC_CP_B128(src, dst, 32);
      ASYNC_CP_B128(src, dst, 48);
#else
      *(v8bf*)(dst + 0)  = *(const v8bf*)(src + 0);
      *(v8bf*)(dst + 8)  = *(const v8bf*)(src + 8);
      *(v8bf*)(dst + 16) = *(const v8bf*)(src + 16);
      *(v8bf*)(dst + 24) = *(const v8bf*)(src + 24);
#endif
    }
    {
      const float* s0 = W + (size_t)(kt + 2 * wkp) * N + n0 + wns;
      const float* s1 = s0 + N;
      if (kt + BKt < K) __builtin_prefetch(s0 + (size_t)BKt * N, 0, 1);
      #pragma unroll
      for (int j = 0; j < 16; ++j) {
        const bool ok = (n0 + wns + j) < N;
        const float w0 = ok ? s0[j] : 0.0f;
        const float w1 = ok ? s1[j] : 0.0f;
        *(unsigned*)&sB[(wns + j) * BKt + 2 * wkp] = pack2f(w0, w1);
      }
    }
#if USE_ASYNC_LDS
    WAIT_ASYNC();
#endif
    __syncthreads();
    #pragma unroll
    for (int kk = 0; kk < BKt; kk += 32) {
      v16bf af[2], bfv[4];
      #pragma unroll
      for (int i = 0; i < 2; ++i)
        af[i] = load_frag(&sA[(wr * 32 + i * 16 + lm) * BKt + kk + hb * 8]);
      #pragma unroll
      for (int j = 0; j < 4; ++j)
        bfv[j] = load_frag(&sB[(wc * 64 + j * 16 + lm) * BKt + kk + hb * 8]);
      #pragma unroll
      for (int i = 0; i < 2; ++i)
        #pragma unroll
        for (int j = 0; j < 4; ++j)
          acc[i][j] = __builtin_amdgcn_wmma_f32_16x16x32_bf16(
              false, af[i], false, bfv[j], (short)0, acc[i][j], false, false);
    }
    __syncthreads();
  }

  #pragma unroll
  for (int i = 0; i < 2; ++i) {
    #pragma unroll
    for (int j = 0; j < 4; ++j) {
      const int col = n0 + wc * 64 + j * 16 + lm;
      if (col >= N) continue;
      const float bv = BIAS ? bias[col] : 0.0f;
      #pragma unroll
      for (int r = 0; r < 8; ++r) {
        const int row = m0 + wr * 32 + i * 16 + hb * 8 + r;
        if (row >= M) continue;
        float vl = acc[i][j][r] + bv;
        if (GELU) vl = 0.5f * vl * (1.0f + erff(vl * 0.70710678118f));
        if (RES)  vl += resid[(size_t)row * N + col];
        if (OUTBF) outB[(size_t)row * N + col] = f2bf(vl);
        else       outF[(size_t)row * N + col] = vl;
      }
    }
  }
}

// ------------------------------------------------------- flash attention
// Block: 128 queries of one (b,h); 8 waves, wave = 16-query M-tile.
// Loop 64-key tiles: S=q.k^T (WMMA), causal+online softmax, O += P.V (WMMA).
__global__ __launch_bounds__(256) void attention_kernel(
    const bf16_t* __restrict__ q, const bf16_t* __restrict__ k,
    const bf16_t* __restrict__ v, bf16_t* __restrict__ o, float scale) {
  __shared__ bf16_t sK[64 * 64];             // [key][d]
  __shared__ bf16_t sV[64 * 64];             // transposed [d][key]
  __shared__ bf16_t sP[8 * 16 * 64];         // per-wave P staging

  const int tid = threadIdx.x, lane = tid & 31, wave = tid >> 5;
  const int lm = lane & 15, hb = lane >> 4;
  const int q0 = blockIdx.x * 128;
  const int bh = blockIdx.y;
  const int b_ = bh / Hk, h_ = bh % Hk;
  const size_t base = (size_t)b_ * Tk * Ck + (size_t)h_ * Dk;

  v16bf aq[2];
  {
    const bf16_t* qp = q + base + (size_t)(q0 + wave * 16 + lm) * Ck + hb * 8;
    aq[0] = load_frag(qp);
    aq[1] = load_frag(qp + 32);
  }

  const v8f zv = {};
  v8f accO[4];
  #pragma unroll
  for (int dt = 0; dt < 4; ++dt) accO[dt] = zv;
  float rmax[8], rsum[8];
  #pragma unroll
  for (int r = 0; r < 8; ++r) { rmax[r] = -1e30f; rsum[r] = 0.f; }

  const int kkey = tid >> 2, kds = (tid & 3) * 16;   // sK: 16 bf16 per thread
  const int vkp  = tid >> 3, vds = (tid & 7) * 8;    // sV: key-pair x 8 d per thread

  const int ktiles = (q0 + 128) >> 6;
  for (int kt = 0; kt < ktiles; ++kt) {
    const int kb = kt * 64;
    __syncthreads();
    {
      // K tile: straight byte copy -> async engine when available
      const bf16_t* kp_ = k + base + (size_t)(kb + kkey) * Ck + kds;
      bf16_t* dK = &sK[kkey * 64 + kds];
#if USE_ASYNC_LDS
      ASYNC_CP_B128(kp_, dK, 0);
      ASYNC_CP_B128(kp_, dK, 16);
#else
      *(v8bf*)(dK + 0) = *(const v8bf*)(kp_ + 0);
      *(v8bf*)(dK + 8) = *(const v8bf*)(kp_ + 8);
#endif
      // V tile: transpose [key][d] -> [d][key], packing key pairs into u32
      const bf16_t* vp0 = v + base + (size_t)(kb + 2 * vkp) * Ck + vds;
      const bf16_t* vp1 = vp0 + Ck;
      const v8bf a0 = *(const v8bf*)vp0;
      const v8bf a1 = *(const v8bf*)vp1;
      #pragma unroll
      for (int j = 0; j < 8; ++j)
        *(unsigned*)&sV[(vds + j) * 64 + 2 * vkp] = packbf(a0[j], a1[j]);
    }
#if USE_ASYNC_LDS
    WAIT_ASYNC();
#endif
    __syncthreads();

    v8f s[4];
    #pragma unroll
    for (int j = 0; j < 4; ++j) s[j] = zv;
    #pragma unroll
    for (int ss = 0; ss < 2; ++ss)
      #pragma unroll
      for (int j = 0; j < 4; ++j) {
        v16bf bk = load_frag(&sK[(j * 16 + lm) * 64 + ss * 32 + hb * 8]);
        s[j] = __builtin_amdgcn_wmma_f32_16x16x32_bf16(
            false, aq[ss], false, bk, (short)0, s[j], false, false);
      }

    float tmax[8];
    #pragma unroll
    for (int r = 0; r < 8; ++r) tmax[r] = -1e30f;
    #pragma unroll
    for (int j = 0; j < 4; ++j) {
      const int kidx = kb + j * 16 + lm;
      #pragma unroll
      for (int r = 0; r < 8; ++r) {
        const int qidx = q0 + wave * 16 + hb * 8 + r;
        float val = s[j][r] * scale;
        if (kidx > qidx) val = -1e30f;
        s[j][r] = val;
        tmax[r] = fmaxf(tmax[r], val);
      }
    }
    #pragma unroll
    for (int r = 0; r < 8; ++r)
      #pragma unroll
      for (int off = 1; off < 16; off <<= 1)
        tmax[r] = fmaxf(tmax[r], __shfl_xor(tmax[r], off));

    float alpha[8], psum[8];
    #pragma unroll
    for (int r = 0; r < 8; ++r) {
      const float nm = fmaxf(rmax[r], tmax[r]);
      alpha[r] = __expf(rmax[r] - nm);
      rmax[r] = nm;
      psum[r] = 0.f;
    }
    #pragma unroll
    for (int j = 0; j < 4; ++j)
      #pragma unroll
      for (int r = 0; r < 8; ++r) {
        const float pp = __expf(s[j][r] - rmax[r]);
        s[j][r] = pp;
        psum[r] += pp;
      }
    #pragma unroll
    for (int r = 0; r < 8; ++r) {
      #pragma unroll
      for (int off = 1; off < 16; off <<= 1) psum[r] += __shfl_xor(psum[r], off);
      rsum[r] = rsum[r] * alpha[r] + psum[r];
    }
    #pragma unroll
    for (int dt = 0; dt < 4; ++dt)
      #pragma unroll
      for (int r = 0; r < 8; ++r) accO[dt][r] *= alpha[r];

    bf16_t* pw = &sP[wave * 1024];           // same-wave LDS round trip
    #pragma unroll
    for (int j = 0; j < 4; ++j)
      #pragma unroll
      for (int r = 0; r < 8; ++r)
        pw[(hb * 8 + r) * 64 + j * 16 + lm] = f2bf(s[j][r]);

    #pragma unroll
    for (int ss = 0; ss < 2; ++ss) {
      v16bf ap = load_frag(&pw[lm * 64 + ss * 32 + hb * 8]);
      #pragma unroll
      for (int dt = 0; dt < 4; ++dt) {
        v16bf bv = load_frag(&sV[(dt * 16 + lm) * 64 + ss * 32 + hb * 8]);
        accO[dt] = __builtin_amdgcn_wmma_f32_16x16x32_bf16(
            false, ap, false, bv, (short)0, accO[dt], false, false);
      }
    }
  }

  #pragma unroll
  for (int dt = 0; dt < 4; ++dt)
    #pragma unroll
    for (int r = 0; r < 8; ++r) {
      const int row = q0 + wave * 16 + hb * 8 + r;
      o[base + (size_t)row * Ck + dt * 16 + lm] = f2bf(accO[dt][r] / rsum[r]);
    }
}

// ---------------------------------------------------------------- driver
extern "C" void kernel_launch(void* const* d_in, const int* in_sizes, int n_in,
                              void* d_out, int out_size, void* d_ws, size_t ws_size,
                              hipStream_t stream) {
  (void)in_sizes; (void)n_in; (void)out_size; (void)ws_size;
  const int*   tokens = (const int*)  d_in[0];
  const float* wte    = (const float*)d_in[1];
  const float* wpe    = (const float*)d_in[2];
  const float* Wq     = (const float*)d_in[3];
  const float* Wk     = (const float*)d_in[4];
  const float* Wv     = (const float*)d_in[5];
  const float* Wo     = (const float*)d_in[6];
  const float* bo     = (const float*)d_in[7];
  const float* ln1g   = (const float*)d_in[8];
  const float* ln1b   = (const float*)d_in[9];
  const float* ln2g   = (const float*)d_in[10];
  const float* ln2b   = (const float*)d_in[11];
  const float* W1     = (const float*)d_in[12];
  const float* b1     = (const float*)d_in[13];
  const float* W2     = (const float*)d_in[14];
  const float* b2     = (const float*)d_in[15];
  const float* lnfg   = (const float*)d_in[16];
  const float* lnfb   = (const float*)d_in[17];
  const float* Wh     = (const float*)d_in[18];
  const float* bh     = (const float*)d_in[19];

  const int M = Bk * Tk;                     // 4096 rows
  char* p = (char*)d_ws;
  float*  hbuf = (float*)p;  p += (size_t)M * Ck * sizeof(float);
  bf16_t* xb   = (bf16_t*)p; p += (size_t)M * Ck * sizeof(bf16_t);
  bf16_t* qb   = (bf16_t*)p; p += (size_t)M * Ck * sizeof(bf16_t);
  bf16_t* kb   = (bf16_t*)p; p += (size_t)M * Ck * sizeof(bf16_t);
  bf16_t* vb   = (bf16_t*)p; p += (size_t)M * Ck * sizeof(bf16_t);
  bf16_t* ob   = (bf16_t*)p; p += (size_t)M * Ck * sizeof(bf16_t);
  bf16_t* mb   = (bf16_t*)p; p += (size_t)M * 4 * Ck * sizeof(bf16_t);

  embed_kernel<<<M, 256, 0, stream>>>(tokens, wte, wpe, hbuf);

  const dim3 blk(256);
  const dim3 gP(Ck / 128, M / 128);          // 6 x 32
  const dim3 gM1(4 * Ck / 128, M / 128);     // 24 x 32
  const dim3 gAtt(Tk / 128, Bk * Hk);        // 8 x 48
  const float scale = 1.0f / sqrtf((float)Ck);   // reference uses sqrt(C)

  for (int l = 0; l < Lk; ++l) {
    const size_t wo = (size_t)l * Ck * Ck;
    layernorm_kernel<<<M, blk, 0, stream>>>(hbuf, ln1g + l * Ck, ln1b + l * Ck, xb);
    gemm_bf16_kernel<false,false,false,true><<<gP, blk, 0, stream>>>(
        xb, Wq + wo, nullptr, nullptr, nullptr, qb, M, Ck, Ck);
    gemm_bf16_kernel<false,false,false,true><<<gP, blk, 0, stream>>>(
        xb, Wk + wo, nullptr, nullptr, nullptr, kb, M, Ck, Ck);
    gemm_bf16_kernel<false,false,false,true><<<gP, blk, 0, stream>>>(
        xb, Wv + wo, nullptr, nullptr, nullptr, vb, M, Ck, Ck);
    attention_kernel<<<gAtt, blk, 0, stream>>>(qb, kb, vb, ob, scale);
    gemm_bf16_kernel<true,false,true,false><<<gP, blk, 0, stream>>>(
        ob, Wo + wo, bo + l * Ck, hbuf, hbuf, nullptr, M, Ck, Ck);
    layernorm_kernel<<<M, blk, 0, stream>>>(hbuf, ln2g + l * Ck, ln2b + l * Ck, xb);
    gemm_bf16_kernel<true,true,false,true><<<gM1, blk, 0, stream>>>(
        xb, W1 + (size_t)l * Ck * 4 * Ck, b1 + (size_t)l * 4 * Ck,
        nullptr, nullptr, mb, M, 4 * Ck, Ck);
    gemm_bf16_kernel<true,false,true,false><<<gP, blk, 0, stream>>>(
        mb, W2 + (size_t)l * 4 * Ck * Ck, b2 + l * Ck, hbuf, hbuf, nullptr,
        M, Ck, 4 * Ck);
  }
  layernorm_kernel<<<M, blk, 0, stream>>>(hbuf, lnfg, lnfb, xb);
  const dim3 gH((Vk + 127) / 128, M / 128);
  gemm_bf16_kernel<true,false,false,false><<<gH, blk, 0, stream>>>(
      xb, Wh, bh, nullptr, (float*)d_out, nullptr, M, Vk, Ck);
}
